// MiniDrift_88270167868072
// MI455X (gfx1250) — matrix operand order repaired and checked by reading, not compiled, not run on hardware
//
#include <hip/hip_runtime.h>
#include <hip/hip_bf16.h>
#include <math.h>

typedef __bf16 bf16x16 __attribute__((ext_vector_type(16)));
typedef float  f32x8   __attribute__((ext_vector_type(8)));

#define TPB 256

__device__ __forceinline__ unsigned f2bf1(float f) {
  unsigned u = __float_as_uint(f);
  return (u + 0x7FFFu + ((u >> 16) & 1u)) >> 16;   // round-to-nearest-even bf16
}
__device__ __forceinline__ unsigned packbf(float lo, float hi) {
  return f2bf1(lo) | (f2bf1(hi) << 16);
}

// One GEMM template for the whole pipeline.
//   C(M,N) = A(M,K) * op(B) + epilogue
// BLAY: 0 = B is (N,K) row-major ("NT", weights / gram), 1 = B is (K,N) row-major ("NN")
// ATR : 0 = raw A; 1 = A := exp(A - 0.5*(rlse[row] + clse[colofs + k]))  (soft matrix rebuild)
// MODE: 0 = acc + bias[n]
//       1 = res[row][n] + relu(acc + bias[n])            (residual block)
//       2 = -sqrt(max(p[row]+q[col]-2*acc,0))/T  (-1e6/T on diag if diag)   T=0.2
//       3 = plain acc
template<int BLAY, int ATR, int MODE>
__global__ __launch_bounds__(TPB)
void gemm_wmma(const float* __restrict__ A, int lda,
               const float* __restrict__ B, int ldb,
               float* __restrict__ C, int ldc,
               int K,
               const float* __restrict__ bias,
               const float* __restrict__ res, int ldres,
               const float* __restrict__ p, const float* __restrict__ q,
               const float* __restrict__ rlse, const float* __restrict__ clse,
               int colofs, int diag)
{
  // bf16 tiles, row stride 20 dwords (40 halves) -> 16B-aligned rows, conflict-friendly
  __shared__ unsigned As[128 * 20];
  __shared__ unsigned Bs[128 * 20];

  const int tid  = threadIdx.x;
  const int row0 = blockIdx.y * 128;
  const int col0 = blockIdx.x * 128;

  const int wid  = tid >> 5;
  const int lane = tid & 31;
  const int lr   = lane & 15;
  const int lh   = lane >> 4;
  const int wr   = wid >> 2;   // 0..1 : 64-row slab
  const int wc   = wid & 3;    // 0..3 : 32-col slab

  f32x8 acc[4][2];
  for (int mi = 0; mi < 4; ++mi)
    for (int ni = 0; ni < 2; ++ni)
      for (int e = 0; e < 8; ++e) acc[mi][ni][e] = 0.0f;

  for (int k0 = 0; k0 < K; k0 += 32) {
    __syncthreads();
    { // stage A: 128x32 f32 -> bf16 (optionally rebuilding soft weights)
      int r  = tid >> 3;
      int c4 = (tid & 7) << 2;
      for (int s = 0; s < 4; ++s, r += 32) {
        const float* src = A + (size_t)(row0 + r) * lda + (k0 + c4);
        float4 v = *(const float4*)src;
        if (ATR) {
          float rb = 0.5f * rlse[row0 + r];
          int cb = colofs + k0 + c4;
          v.x = __expf(v.x - rb - 0.5f * clse[cb + 0]);
          v.y = __expf(v.y - rb - 0.5f * clse[cb + 1]);
          v.z = __expf(v.z - rb - 0.5f * clse[cb + 2]);
          v.w = __expf(v.w - rb - 0.5f * clse[cb + 3]);
        }
        As[r * 20 + (c4 >> 1) + 0] = packbf(v.x, v.y);
        As[r * 20 + (c4 >> 1) + 1] = packbf(v.z, v.w);
      }
    }
    if (BLAY == 0) { // B is (N,K): same k-contiguous staging
      int r  = tid >> 3;
      int c4 = (tid & 7) << 2;
      for (int s = 0; s < 4; ++s, r += 32) {
        const float* src = B + (size_t)(col0 + r) * ldb + (k0 + c4);
        float4 v = *(const float4*)src;
        Bs[r * 20 + (c4 >> 1) + 0] = packbf(v.x, v.y);
        Bs[r * 20 + (c4 >> 1) + 1] = packbf(v.z, v.w);
      }
    } else {        // B is (K,N): coalesced read along n, transposed scatter into LDS
      unsigned short* Bsh = (unsigned short*)Bs;
      int k  = tid >> 5;
      int n4 = (tid & 31) << 2;
      for (int s = 0; s < 4; ++s, k += 8) {
        const float* src = B + (size_t)(k0 + k) * ldb + (col0 + n4);
        float4 v = *(const float4*)src;
        Bsh[(n4 + 0) * 40 + k] = (unsigned short)f2bf1(v.x);
        Bsh[(n4 + 1) * 40 + k] = (unsigned short)f2bf1(v.y);
        Bsh[(n4 + 2) * 40 + k] = (unsigned short)f2bf1(v.z);
        Bsh[(n4 + 3) * 40 + k] = (unsigned short)f2bf1(v.w);
      }
    }
    __syncthreads();

    // A fragment (16x32): lane<16 row=lr K0..15; lane>=16 K16..31
    union { unsigned u[8]; bf16x16 v; } af[4], bfr[2];
    for (int mi = 0; mi < 4; ++mi) {
      const unsigned* pA = &As[(wr * 64 + mi * 16 + lr) * 20];
      for (int i = 0; i < 8; ++i) af[mi].u[i] = pA[8 * lh + i];
    }
    // B fragment (32x16): lane<16 col=lr K0..15; lane>=16 K16..31
    for (int ni = 0; ni < 2; ++ni) {
      const unsigned* pB = &Bs[(wc * 32 + ni * 16 + lr) * 20 + 8 * lh];
      for (int i = 0; i < 8; ++i) bfr[ni].u[i] = pB[i];
    }
    for (int mi = 0; mi < 4; ++mi)
      for (int ni = 0; ni < 2; ++ni)
        acc[mi][ni] = __builtin_amdgcn_wmma_f32_16x16x32_bf16(
            false, af[mi].v, false, bfr[ni].v, (short)0, acc[mi][ni], false, false);
  }

  // Epilogue: C layout — VGPR e: row = e + 8*lh, col = lr
  for (int mi = 0; mi < 4; ++mi) {
    for (int ni = 0; ni < 2; ++ni) {
      int colL = col0 + wc * 32 + ni * 16 + lr;
      for (int e = 0; e < 8; ++e) {
        int rowL = row0 + wr * 64 + mi * 16 + 8 * lh + e;
        float v = acc[mi][ni][e];
        float out;
        if (MODE == 0) {
          out = v + bias[colL];
        } else if (MODE == 1) {
          float t = v + bias[colL];
          out = res[(size_t)rowL * ldres + colL] + fmaxf(t, 0.0f);
        } else if (MODE == 2) {
          float d2 = fmaxf(p[rowL] + q[colL] - 2.0f * v, 0.0f);
          out = -5.0f * sqrtf(d2);              // -d / T, T = 0.2
          if (diag && rowL == colL) out -= 5.0e6f;
        } else {
          out = v;
        }
        C[(size_t)rowL * ldc + colL] = out;
      }
    }
  }
}

// ---- reductions ------------------------------------------------------------

__global__ __launch_bounds__(128) void row_norm2_128(const float* __restrict__ X,
                                                     float* __restrict__ out) {
  int row = blockIdx.x;
  float v = X[(size_t)row * 128 + threadIdx.x];
  float s = v * v;
  for (int o = 16; o > 0; o >>= 1) s += __shfl_down(s, o, 32);
  __shared__ float red[4];
  if ((threadIdx.x & 31) == 0) red[threadIdx.x >> 5] = s;
  __syncthreads();
  if (threadIdx.x == 0) out[row] = red[0] + red[1] + red[2] + red[3];
}

__global__ __launch_bounds__(256) void row_lse_k(const float* __restrict__ Abig,
                                                 float* __restrict__ rlse) {
  int row = blockIdx.x, tid = threadIdx.x;
  const float* a = Abig + (size_t)row * 8192;
  __shared__ float red[256];
  float m = -3.0e38f;
  for (int j = tid; j < 8192; j += 256) m = fmaxf(m, a[j]);
  red[tid] = m; __syncthreads();
  for (int o = 128; o > 0; o >>= 1) { if (tid < o) red[tid] = fmaxf(red[tid], red[tid + o]); __syncthreads(); }
  m = red[0]; __syncthreads();
  float s = 0.0f;
  for (int j = tid; j < 8192; j += 256) s += __expf(a[j] - m);
  red[tid] = s; __syncthreads();
  for (int o = 128; o > 0; o >>= 1) { if (tid < o) red[tid] += red[tid + o]; __syncthreads(); }
  if (tid == 0) rlse[row] = m + __logf(red[0]);
}

__global__ __launch_bounds__(256) void col_lse_k(const float* __restrict__ Abig,
                                                 float* __restrict__ clse) {
  int col = blockIdx.x * 256 + threadIdx.x;
  const float* a = Abig + col;
  float m = -3.0e38f, s = 0.0f;                 // online max/sum down the column
  for (int r = 0; r < 4096; ++r) {
    float v = a[(size_t)r * 8192];
    if (v > m) { s = s * __expf(m - v) + 1.0f; m = v; }
    else       { s += __expf(v - m); }
  }
  clse[col] = m + __logf(s);
}

__global__ __launch_bounds__(256) void rowsum_k(const float* __restrict__ Abig,
    const float* __restrict__ rlse, const float* __restrict__ clse,
    float* __restrict__ s_pos, float* __restrict__ s_neg) {
  int row = blockIdx.x, tid = threadIdx.x;
  const float* a = Abig + (size_t)row * 8192;
  float hr = 0.5f * rlse[row];
  float sp = 0.0f, sn = 0.0f;
  for (int j = tid;        j < 4096; j += 256) sp += __expf(a[j] - hr - 0.5f * clse[j]);
  for (int j = 4096 + tid; j < 8192; j += 256) sn += __expf(a[j] - hr - 0.5f * clse[j]);
  __shared__ float rp[256], rn[256];
  rp[tid] = sp; rn[tid] = sn; __syncthreads();
  for (int o = 128; o > 0; o >>= 1) {
    if (tid < o) { rp[tid] += rp[tid + o]; rn[tid] += rn[tid + o]; }
    __syncthreads();
  }
  if (tid == 0) { s_pos[row] = rp[0]; s_neg[row] = rn[0]; }
}

// loss = mean( (s_neg[i]*U1 - s_pos[i]*s_neg[i]*U2)^2 ) — deterministic 2-stage
__global__ __launch_bounds__(256) void loss_partial_k(const float* __restrict__ U1,
    const float* __restrict__ U2, const float* __restrict__ s_pos,
    const float* __restrict__ s_neg, float* __restrict__ part) {
  float acc = 0.0f;
  for (int idx = blockIdx.x * 256 + threadIdx.x; idx < 4096 * 128; idx += gridDim.x * 256) {
    int i = idx >> 7;
    float sn = s_neg[i];
    float v = sn * U1[idx] - s_pos[i] * sn * U2[idx];
    acc += v * v;
  }
  __shared__ float red[256];
  int tid = threadIdx.x;
  red[tid] = acc; __syncthreads();
  for (int o = 128; o > 0; o >>= 1) { if (tid < o) red[tid] += red[tid + o]; __syncthreads(); }
  if (tid == 0) part[blockIdx.x] = red[0];
}

__global__ __launch_bounds__(256) void loss_final_k(const float* __restrict__ part, int n,
                                                    float* __restrict__ out) {
  __shared__ float red[256];
  float acc = 0.0f;
  for (int i = threadIdx.x; i < n; i += 256) acc += part[i];
  red[threadIdx.x] = acc; __syncthreads();
  for (int o = 128; o > 0; o >>= 1) { if (threadIdx.x < o) red[threadIdx.x] += red[threadIdx.x + o]; __syncthreads(); }
  if (threadIdx.x == 0) out[0] = red[0] * (1.0f / (4096.0f * 128.0f));
}

// ---- driver ----------------------------------------------------------------

extern "C" void kernel_launch(void* const* d_in, const int* in_sizes, int n_in,
                              void* d_out, int out_size, void* d_ws, size_t ws_size,
                              hipStream_t stream) {
  const float* y_pos = (const float*)d_in[0];
  const float* eps   = (const float*)d_in[1];
  const float* W_in  = (const float*)d_in[2];
  const float* b_in  = (const float*)d_in[3];
  const float* W_blk = (const float*)d_in[4];
  const float* b_blk = (const float*)d_in[5];
  const float* W_out = (const float*)d_in[6];
  const float* b_out = (const float*)d_in[7];

  const int N = 4096, D = 128, H = 512;
  float* ws = (float*)d_ws;
  size_t off = 0;
  float* Abig = ws + off; off += (size_t)N * 2 * N;   // 4096x8192 logits (128 MB)
  float* xb0  = ws + off; off += (size_t)N * H;
  float* xb1  = ws + off; off += (size_t)N * H;
  float* xout = ws + off; off += (size_t)N * D;
  float* nx   = ws + off; off += N;
  float* ny   = ws + off; off += N;
  float* rlse = ws + off; off += N;
  float* clse = ws + off; off += 2 * N;
  float* spos = ws + off; off += N;
  float* sneg = ws + off; off += N;
  float* part = ws + off; off += 512;
  float* U1 = xb0;                                    // reuse MLP buffers after xout
  float* U2 = xb1;

  dim3 blk(256);
  // Generator MLP
  gemm_wmma<0,0,0><<<dim3(H/128, N/128), blk, 0, stream>>>(eps, D, W_in, D, xb0, H, D,
      b_in, nullptr, 0, nullptr, nullptr, nullptr, nullptr, 0, 0);
  float* cur = xb0; float* nxt = xb1;
  for (int i = 0; i < 4; ++i) {
    gemm_wmma<0,0,1><<<dim3(H/128, N/128), blk, 0, stream>>>(cur, H,
        W_blk + (size_t)i * H * H, H, nxt, H, H,
        b_blk + (size_t)i * H, cur, H, nullptr, nullptr, nullptr, nullptr, 0, 0);
    float* t = cur; cur = nxt; nxt = t;
  }
  gemm_wmma<0,0,0><<<dim3(D/128, N/128), blk, 0, stream>>>(cur, H, W_out, H, xout, D, H,
      b_out, nullptr, 0, nullptr, nullptr, nullptr, nullptr, 0, 0);

  // Squared row norms
  row_norm2_128<<<N, 128, 0, stream>>>(xout, nx);
  row_norm2_128<<<N, 128, 0, stream>>>(y_pos, ny);

  // Logits A = [-dist_pos/T | -dist_neg/T - eye*1e6/T]
  gemm_wmma<0,0,2><<<dim3(N/128, N/128), blk, 0, stream>>>(xout, D, y_pos, D, Abig, 2*N, D,
      nullptr, nullptr, 0, nx, ny, nullptr, nullptr, 0, 0);
  gemm_wmma<0,0,2><<<dim3(N/128, N/128), blk, 0, stream>>>(xout, D, xout, D, Abig + N, 2*N, D,
      nullptr, nullptr, 0, nx, nx, nullptr, nullptr, 0, 1);

  // sqrt(softmax_row * softmax_col) = exp(a - (rlse_i + clse_j)/2)
  row_lse_k<<<N, 256, 0, stream>>>(Abig, rlse);
  col_lse_k<<<(2 * N) / 256, 256, 0, stream>>>(Abig, clse);
  rowsum_k<<<N, 256, 0, stream>>>(Abig, rlse, clse, spos, sneg);

  // U1 = A_soft_pos @ y_pos ; U2 = A_soft_neg @ xout  (weights rebuilt in staging)
  gemm_wmma<1,1,3><<<dim3(D/128, N/128), blk, 0, stream>>>(Abig, 2*N, y_pos, D, U1, D, N,
      nullptr, nullptr, 0, nullptr, nullptr, rlse, clse, 0, 0);
  gemm_wmma<1,1,3><<<dim3(D/128, N/128), blk, 0, stream>>>(Abig + N, 2*N, xout, D, U2, D, N,
      nullptr, nullptr, 0, nullptr, nullptr, rlse, clse, N, 0);

  // loss = mean(V^2), V = s_neg*U1 - s_pos*s_neg*U2
  loss_partial_k<<<512, 256, 0, stream>>>(U1, U2, spos, sneg, part);
  loss_final_k<<<1, 256, 0, stream>>>(part, 512, (float*)d_out);
}
